// GraphVAE_7069516169733
// MI455X (gfx1250) — matrix operand order repaired and checked by reading, not compiled, not run on hardware
//
#include <hip/hip_runtime.h>
#include <hip/hip_bf16.h>

// ---------------- dims ----------------
#define MM 4
#define BB 256
#define KK 64
#define HH 1024
#define NN 50
#define FD 3
#define PP 64
#define EE 1225      // N*(N-1)/2
#define E_PAD 1280   // 80*16  (divisible by 64 for 4-tile wave blocking)
#define F_PAD 192    // 12*16
#define MB (MM*BB)   // 1024 rows in all GEMMs

typedef __attribute__((ext_vector_type(16))) __bf16 v16bf;
typedef __attribute__((ext_vector_type(8)))  __bf16 v8bf;
typedef __attribute__((ext_vector_type(8)))  float  v8f;

__device__ __forceinline__ __bf16 f2bf(float x) { return (__bf16)x; }

__device__ __forceinline__ float softplusf(float x) {
    return fmaxf(x, 0.f) + log1pf(expf(-fabsf(x)));
}
__device__ __forceinline__ float logsigf(float x) {
    return fminf(x, 0.f) - log1pf(expf(-fabsf(x)));
}

// ------------- WMMA: one wave computes 4 adjacent 16x16 tiles (64 output cols) -------------
// A row-major (lda = K stride), B transposed: BT[n*ldk + k] (contiguous K per column)
__device__ __forceinline__ void wmma_tile4(const __bf16* __restrict__ A, int lda,
                                           const __bf16* __restrict__ BT, int ldk,
                                           int Ktot, int row0, int col0, int lane,
                                           v8f acc[4]) {
    const int half  = lane >> 4;           // 0 or 1
    const int r     = row0 + (lane & 15);  // A row this lane owns
    const int nbase = col0 + (lane & 15);  // first B/C column this lane owns
    for (int k0 = 0; k0 < Ktot; k0 += 32) {
        // A 16x32, contiguous k order used here (same as B): lane half 0 -> K {0..15}, half 1 -> K {16..31}
        const __bf16* ap = A + (size_t)r * lda + k0 + half * 16;
        v8bf lo = *(const v8bf*)(ap);
        v8bf hi = *(const v8bf*)(ap + 8);
        v16bf a = __builtin_shufflevector(lo, hi, 0, 1, 2, 3, 4, 5, 6, 7,
                                                  8, 9, 10, 11, 12, 13, 14, 15);
#pragma unroll
        for (int t = 0; t < 4; ++t) {
            // B 32x16, same contiguous k order as A: lane half selects K 0..15 vs 16..31 -> contiguous in BT
            v16bf b = *(const v16bf*)(BT + (size_t)(nbase + 16 * t) * ldk + k0 + 16 * half);
            acc[t] = __builtin_amdgcn_wmma_f32_16x16x32_bf16(false, a, false, b,
                                                             (short)0, acc[t], false, false);
        }
    }
}

// ---------------- elementwise / conversion kernels ----------------
__global__ void zb_kernel(const float* __restrict__ mean, const float* __restrict__ logvar,
                          const float* __restrict__ eps, __bf16* __restrict__ Zb) {
    int idx = blockIdx.x * 256 + threadIdx.x;
    if (idx >= MM * BB * KK) return;
    int k = idx & (KK - 1);
    int b = (idx >> 6) & (BB - 1);
    float z = mean[b * KK + k] + expf(0.5f * logvar[b * KK + k]) * eps[idx];
    Zb[idx] = f2bf(z);
}

// transpose + pad + f32->bf16: dst[n*ldk + k] = src[k*cols + n] (n < cols else 0)
__global__ void tcvt_kernel(const float* __restrict__ src, __bf16* __restrict__ dst,
                            int cols, int ldk, int total) {
    int idx = blockIdx.x * 256 + threadIdx.x;
    if (idx >= total) return;
    int n = idx / ldk, k = idx - n * ldk;
    dst[idx] = f2bf(n < cols ? src[(size_t)k * cols + n] : 0.f);
}

// ---------------- GEMM1: h = relu(Zb @ W1 + b1), bf16 out ----------------
__global__ void gemm_relu_kernel(const __bf16* __restrict__ A, const __bf16* __restrict__ BT,
                                 const float* __restrict__ bias, __bf16* __restrict__ out) {
    int lane = threadIdx.x & 31, wave = threadIdx.x >> 5;
    int group = blockIdx.x * 4 + wave;     // 16 groups of 4 col tiles, grid.x = 4
    int row0 = blockIdx.y * 16;
    int col0 = group * 64;
    v8f acc[4] = {};
    wmma_tile4(A, KK, BT, KK, KK, row0, col0, lane, acc);
    int half = lane >> 4;
#pragma unroll
    for (int t = 0; t < 4; ++t) {
        int n = col0 + 16 * t + (lane & 15);
        float bs = bias[n];
#pragma unroll
        for (int r = 0; r < 8; ++r) {
            float v = acc[t][r] + bs;
            out[(size_t)(row0 + r + 8 * half) * HH + n] = f2bf(v > 0.f ? v : 0.f);
        }
    }
}

// ---------------- GEMM2/3: f32 out + bias (zero bias in padded cols) ----------------
__global__ void gemm_f32_kernel(const __bf16* __restrict__ A, const __bf16* __restrict__ BT,
                                const float* __restrict__ bias, float* __restrict__ out,
                                int ldb, int ngroups, int validcols) {
    int lane = threadIdx.x & 31, wave = threadIdx.x >> 5;
    int group = blockIdx.x * 4 + wave;
    if (group >= ngroups) return;
    int row0 = blockIdx.y * 16;
    int col0 = group * 64;
    v8f acc[4] = {};
    wmma_tile4(A, HH, BT, HH, HH, row0, col0, lane, acc);
    int half = lane >> 4;
#pragma unroll
    for (int t = 0; t < 4; ++t) {
        int n = col0 + 16 * t + (lane & 15);
        float bs = (n < validcols) ? bias[n] : 0.f;
#pragma unroll
        for (int r = 0; r < 8; ++r)
            out[(size_t)(row0 + r + 8 * half) * ldb + n] = acc[t][r] + bs;
    }
}

// ---------------- Lbar = mean_m logits_A ; Sbar = mean_m sum_e softplus ----------------
__global__ void lbar_kernel(const float* __restrict__ logA, float* __restrict__ Lbar,
                            float* __restrict__ Sbar) {
    int b = blockIdx.x;
    float s = 0.f;
    for (int e = threadIdx.x; e < EE; e += 256) {
        float sum = 0.f;
#pragma unroll
        for (int m = 0; m < MM; ++m) {
            float l = logA[(size_t)(m * BB + b) * E_PAD + e];
            sum += l;
            s += softplusf(l);
        }
        Lbar[(size_t)b * E_PAD + e] = 0.25f * sum;
    }
#pragma unroll
    for (int off = 16; off; off >>= 1) s += __shfl_xor(s, off, 32);
    __shared__ float red[8];
    if ((threadIdx.x & 31) == 0) red[threadIdx.x >> 5] = s;
    __syncthreads();
    if (threadIdx.x == 0) {
        float t = 0.f;
        for (int i = 0; i < 8; ++i) t += red[i];
        Sbar[b] = 0.25f * t;
    }
}

// ---------------- Fx term: coeff . [mean_m lp_bern(2), mean_m lp_cat] ----------------
__global__ void fx_kernel(const float* __restrict__ logF, const float* __restrict__ Fx,
                          const float* __restrict__ coeff, float* __restrict__ fxval) {
    int b = blockIdx.x;
    int lane = threadIdx.x & 31;
    int m = threadIdx.x >> 5;  // blockDim = 128 -> m in 0..3
    const float* L = logF + (size_t)(m * BB + b) * F_PAD;
    const float* F = Fx + (size_t)b * NN * FD;
    float mx = -3.0e38f;
    for (int n = lane; n < NN; n += 32) mx = fmaxf(mx, L[n * 3 + 2]);
#pragma unroll
    for (int off = 16; off; off >>= 1) mx = fmaxf(mx, __shfl_xor(mx, off, 32));
    float se = 0.f;
    for (int n = lane; n < NN; n += 32) se += expf(L[n * 3 + 2] - mx);
#pragma unroll
    for (int off = 16; off; off >>= 1) se += __shfl_xor(se, off, 32);
    float lse = mx + logf(se);
    float c0 = coeff[0], c1 = coeff[1], c2 = coeff[2];
    float acc = 0.f;
    for (int n = lane; n < NN; n += 32) {
        float l0 = L[n * 3 + 0], l1 = L[n * 3 + 1], l2 = L[n * 3 + 2];
        float x0 = F[n * 3 + 0], x1 = F[n * 3 + 1], x2 = F[n * 3 + 2];
        acc += c0 * (x0 * logsigf(l0) + (1.f - x0) * logsigf(-l0));
        acc += c1 * (x1 * logsigf(l1) + (1.f - x1) * logsigf(-l1));
        acc += c2 * (x2 * (l2 - lse));
    }
#pragma unroll
    for (int off = 16; off; off >>= 1) acc += __shfl_xor(acc, off, 32);
    __shared__ float red[4];
    if (lane == 0) red[m] = acc;
    __syncthreads();
    if (threadIdx.x == 0) fxval[b] = 0.25f * (red[0] + red[1] + red[2] + red[3]);
}

// ---------------- KLD ----------------
__global__ void kld_kernel(const float* __restrict__ mean, const float* __restrict__ logvar,
                           float* __restrict__ kldv) {
    int b = blockIdx.x, t = threadIdx.x;  // blockDim = 64
    float mu = mean[b * KK + t], lv = logvar[b * KK + t];
    float v = 0.5f * (expf(lv) + mu * mu - 1.f - lv);
#pragma unroll
    for (int off = 16; off; off >>= 1) v += __shfl_xor(v, off, 32);
    __shared__ float red[2];
    if ((t & 31) == 0) red[t >> 5] = v;
    __syncthreads();
    if (t == 0) kldv[b] = red[0] + red[1];
}

// ---------------- packed tril index table: (i<<8)|j ----------------
__global__ void tril_kernel(int* __restrict__ tij) {
    int e = blockIdx.x * 256 + threadIdx.x;
    if (e >= EE) return;
    int i = (int)floorf((1.0f + sqrtf(8.0f * (float)e + 1.0f)) * 0.5f);
    while (i * (i - 1) / 2 > e) --i;
    while ((i + 1) * i / 2 <= e) ++i;
    tij[e] = (i << 8) | (e - i * (i - 1) / 2);
}

// ---------------- nce_A (gather dot, max over perms) + final combine ----------------
__global__ void nceA_kernel(const float* __restrict__ A_in, const int* __restrict__ perms,
                            const int* __restrict__ tij,
                            const float* __restrict__ Lbar, const float* __restrict__ Sbar,
                            const float* __restrict__ fxval, const float* __restrict__ kldv,
                            float* __restrict__ out) {
    __shared__ float AS[NN * NN];
    __shared__ float LS[EE];
    __shared__ int   PS[PP * NN];
    __shared__ int   TS[EE];
    __shared__ float dotS[PP];
    int b = blockIdx.x;
    for (int i = threadIdx.x; i < NN * NN; i += 256) AS[i] = A_in[(size_t)b * NN * NN + i];
    for (int i = threadIdx.x; i < EE; i += 256)      LS[i] = Lbar[(size_t)b * E_PAD + i];
    for (int i = threadIdx.x; i < PP * NN; i += 256) PS[i] = perms[i];
    for (int i = threadIdx.x; i < EE; i += 256)      TS[i] = tij[i];
    __syncthreads();
    int lane = threadIdx.x & 31, wave = threadIdx.x >> 5;
    for (int p = wave; p < PP; p += 8) {
        const int* pp = PS + p * NN;
        float s = 0.f;
        for (int e = lane; e < EE; e += 32) {
            int v = TS[e];
            s += AS[pp[v >> 8] * NN + pp[v & 255]] * LS[e];
        }
#pragma unroll
        for (int off = 16; off; off >>= 1) s += __shfl_xor(s, off, 32);
        if (lane == 0) dotS[p] = s;
    }
    __syncthreads();
    if (threadIdx.x == 0) {
        float mx = dotS[0];
        for (int p = 1; p < PP; ++p) mx = fmaxf(mx, dotS[p]);
        out[b] = mx - Sbar[b] + fxval[b] - kldv[b];
    }
}

// ---------------- host ----------------
extern "C" void kernel_launch(void* const* d_in, const int* in_sizes, int n_in,
                              void* d_out, int out_size, void* d_ws, size_t ws_size,
                              hipStream_t stream) {
    (void)in_sizes; (void)n_in; (void)out_size; (void)ws_size;
    const float* mean   = (const float*)d_in[0];
    const float* logvar = (const float*)d_in[1];
    const float* eps    = (const float*)d_in[2];
    const float* W1     = (const float*)d_in[3];
    const float* b1     = (const float*)d_in[4];
    const float* WA     = (const float*)d_in[5];
    const float* bA     = (const float*)d_in[6];
    const float* WF     = (const float*)d_in[7];
    const float* bF     = (const float*)d_in[8];
    const float* A_in   = (const float*)d_in[9];
    const float* Fx     = (const float*)d_in[10];
    const float* coeff  = (const float*)d_in[11];
    const int*   perms  = (const int*)d_in[12];
    float* out = (float*)d_out;

    char* ws = (char*)d_ws;
    size_t off = 0;
    auto alloc = [&](size_t bytes) -> void* {
        void* p = ws + off;
        off += (bytes + 255) & ~(size_t)255;
        return p;
    };
    __bf16* Zb   = (__bf16*)alloc((size_t)MB * KK * 2);        // A of GEMM1 (row-major)
    __bf16* W1T  = (__bf16*)alloc((size_t)HH * KK * 2);        // B^T of GEMM1: [n][k]
    __bf16* hb   = (__bf16*)alloc((size_t)MB * HH * 2);        // A of GEMM2/3 (row-major)
    __bf16* WAT  = (__bf16*)alloc((size_t)E_PAD * HH * 2);     // B^T of GEMM2: [e][h]
    __bf16* WFT  = (__bf16*)alloc((size_t)F_PAD * HH * 2);     // B^T of GEMM3: [f][h]
    float*  logA = (float*)alloc((size_t)MB * E_PAD * 4);
    float*  logF = (float*)alloc((size_t)MB * F_PAD * 4);
    float*  Lbar = (float*)alloc((size_t)BB * E_PAD * 4);
    float*  Sbar = (float*)alloc(BB * 4);
    float*  fxv  = (float*)alloc(BB * 4);
    float*  kldv = (float*)alloc(BB * 4);
    int*    tij  = (int*)alloc(EE * 4);

    // index tables (independent)
    tril_kernel<<<(EE + 255) / 256, 256, 0, stream>>>(tij);
    // reparameterize + transposed bf16 weight conversions
    zb_kernel<<<(MM * BB * KK) / 256, 256, 0, stream>>>(mean, logvar, eps, Zb);
    tcvt_kernel<<<(HH * KK) / 256, 256, 0, stream>>>(W1, W1T, HH, KK, HH * KK);
    tcvt_kernel<<<(E_PAD * HH) / 256, 256, 0, stream>>>(WA, WAT, EE, HH, E_PAD * HH);
    tcvt_kernel<<<(F_PAD * HH) / 256, 256, 0, stream>>>(WF, WFT, NN * FD, HH, F_PAD * HH);
    // GEMMs (WMMA bf16 -> f32), 4 tiles per wave, 4 waves per block
    gemm_relu_kernel<<<dim3(4, MB / 16), 128, 0, stream>>>(Zb, W1T, b1, hb);
    gemm_f32_kernel<<<dim3(E_PAD / 64 / 4, MB / 16), 128, 0, stream>>>(
        hb, WAT, bA, logA, E_PAD, E_PAD / 64, EE);
    gemm_f32_kernel<<<dim3(1, MB / 16), 128, 0, stream>>>(
        hb, WFT, bF, logF, F_PAD, F_PAD / 64, NN * FD);
    // reductions
    lbar_kernel<<<BB, 256, 0, stream>>>(logA, Lbar, Sbar);
    fx_kernel<<<BB, 128, 0, stream>>>(logF, Fx, coeff, fxv);
    kld_kernel<<<BB, 64, 0, stream>>>(mean, logvar, kldv);
    // permutation NCE + final combine
    nceA_kernel<<<BB, 256, 0, stream>>>(A_in, perms, tij, Lbar, Sbar, fxv, kldv, out);
}
